// Net_6107443494971
// MI455X (gfx1250) — matrix operand order repaired and checked by reading, not compiled, not run on hardware
//
#include <hip/hip_runtime.h>
#include <math.h>

// ---------------------------------------------------------------------------
// Problem constants (from reference)
// ---------------------------------------------------------------------------
#define NGRAPH   128          // B
#define NODES0   128          // N0
#define EPG      512
#define EDGES    (NGRAPH * EPG)   // 65536
#define F_IN     64
#define EDGE_DIM 16
#define EPAD     32           // edge_attr K padded to 32
#define EMB      128
#define HEADS    4
#define HC       (EMB * HEADS)    // 512
#define DENSE    256
#define NMAX     (NGRAPH * NODES0) // 16384

typedef __attribute__((ext_vector_type(16))) _Float16 v16h;
typedef __attribute__((ext_vector_type(8)))  float    v8f;

// ---------------------------------------------------------------------------
// Device helpers
// ---------------------------------------------------------------------------
__device__ __forceinline__ float wave_sum32(float s) {
#pragma unroll
    for (int off = 16; off > 0; off >>= 1) s += __shfl_xor(s, off, 32);
    return s;
}

// monotone float atomic-max (init with -inf)
__device__ __forceinline__ void atomicMaxF(float* addr, float v) {
    if (v >= 0.0f) atomicMax((int*)addr, __float_as_int(v));
    else           atomicMin((unsigned int*)addr, (unsigned int)__float_as_int(v));
}

// 16 contiguous f32 -> v16h  (4x global_load_b128 + packed cvt)
__device__ __forceinline__ v16h load16_f16(const float* __restrict__ p) {
    const float4* q = (const float4*)p;
    float4 u0 = q[0], u1 = q[1], u2 = q[2], u3 = q[3];
    v16h r;
    r[0]  = (_Float16)u0.x; r[1]  = (_Float16)u0.y; r[2]  = (_Float16)u0.z; r[3]  = (_Float16)u0.w;
    r[4]  = (_Float16)u1.x; r[5]  = (_Float16)u1.y; r[6]  = (_Float16)u1.z; r[7]  = (_Float16)u1.w;
    r[8]  = (_Float16)u2.x; r[9]  = (_Float16)u2.y; r[10] = (_Float16)u2.z; r[11] = (_Float16)u2.w;
    r[12] = (_Float16)u3.x; r[13] = (_Float16)u3.y; r[14] = (_Float16)u3.z; r[15] = (_Float16)u3.w;
    return r;
}

// A-fragment (16-bit A 16x32 layout): 8 floats at ka, 8 floats at ka+16
__device__ __forceinline__ v16h load_afrag(const float* __restrict__ rowp, int ka) {
    const float4* ap = (const float4*)(rowp + ka);
    float4 a0 = ap[0], a1 = ap[1];          // K = ka..ka+7
    float4 a2 = ap[4], a3 = ap[5];          // K = ka+16..ka+23
    v16h r;
    r[0]  = (_Float16)a0.x; r[1]  = (_Float16)a0.y; r[2]  = (_Float16)a0.z; r[3]  = (_Float16)a0.w;
    r[4]  = (_Float16)a1.x; r[5]  = (_Float16)a1.y; r[6]  = (_Float16)a1.z; r[7]  = (_Float16)a1.w;
    r[8]  = (_Float16)a2.x; r[9]  = (_Float16)a2.y; r[10] = (_Float16)a2.z; r[11] = (_Float16)a2.w;
    r[12] = (_Float16)a3.x; r[13] = (_Float16)a3.y; r[14] = (_Float16)a3.z; r[15] = (_Float16)a3.w;
    return r;
}

// ---------------------------------------------------------------------------
// WMMA GEMM (NT): C[M,N] = act(A[M,K] @ Wt[N,K]^T + bias)
// A row-major [M,K], Wt row-major [N,K] (pre-transposed weights), K % 32 == 0.
// block = 128 threads = 4 waves (2x2); wave computes a 32x32 C tile with
// 2 A-frags x 2 B-frags -> 4 wmma per k-step (4 b128 loads per wmma).
// block tile = 64x64; grid = (N/64, M/64).
// ---------------------------------------------------------------------------
__global__ void __launch_bounds__(128)
wmma_gemm_nt_kernel(const float* __restrict__ A, const float* __restrict__ Wt,
                    const float* __restrict__ bias, float* __restrict__ C,
                    int M, int N, int K, int relu)
{
    const int lane  = threadIdx.x & 31;
    const int wv    = threadIdx.x >> 5;                 // 0..3
    const int m0    = blockIdx.y * 64 + (wv >> 1) * 32;
    const int n0    = blockIdx.x * 64 + (wv & 1) * 32;
    const int l15   = lane & 15;
    const int lhalf = lane >> 4;                        // 0 or 1

    const float* __restrict__ a0p = A  + (size_t)(m0 + l15)      * K;
    const float* __restrict__ a1p = A  + (size_t)(m0 + 16 + l15) * K;
    const float* __restrict__ b0p = Wt + (size_t)(n0 + l15)      * K;
    const float* __restrict__ b1p = Wt + (size_t)(n0 + 16 + l15) * K;

    v8f acc00 = {}, acc01 = {}, acc10 = {}, acc11 = {};
    for (int k0 = 0; k0 < K; k0 += 32) {
        const int ka = k0 + lhalf * 8;
        const int kb = k0 + lhalf * 16;
        v16h af0 = load_afrag(a0p, ka);
        v16h af1 = load_afrag(a1p, ka);
        v16h bf0 = load_afrag(b0p, ka);
        v16h bf1 = load_afrag(b1p, ka);

        acc00 = __builtin_amdgcn_wmma_f32_16x16x32_f16(false, af0, false, bf0,
                                                       (short)0, acc00, false, false);
        acc01 = __builtin_amdgcn_wmma_f32_16x16x32_f16(false, af0, false, bf1,
                                                       (short)0, acc01, false, false);
        acc10 = __builtin_amdgcn_wmma_f32_16x16x32_f16(false, af1, false, bf0,
                                                       (short)0, acc10, false, false);
        acc11 = __builtin_amdgcn_wmma_f32_16x16x32_f16(false, af1, false, bf1,
                                                       (short)0, acc11, false, false);
    }

    const int c0 = n0 + l15;        // columns of acc*0
    const int c1 = n0 + 16 + l15;   // columns of acc*1
    const float bv0 = bias ? bias[c0] : 0.0f;
    const float bv1 = bias ? bias[c1] : 0.0f;
#pragma unroll
    for (int r = 0; r < 8; ++r) {
        int ma = m0 + lhalf * 8 + r;        // rows of acc0*
        int mb = ma + 16;                   // rows of acc1*
        float v00 = acc00[r] + bv0;
        float v01 = acc01[r] + bv1;
        float v10 = acc10[r] + bv0;
        float v11 = acc11[r] + bv1;
        if (relu) {
            v00 = fmaxf(v00, 0.0f); v01 = fmaxf(v01, 0.0f);
            v10 = fmaxf(v10, 0.0f); v11 = fmaxf(v11, 0.0f);
        }
        C[(size_t)ma * N + c0] = v00;
        C[(size_t)ma * N + c1] = v01;
        C[(size_t)mb * N + c0] = v10;
        C[(size_t)mb * N + c1] = v11;
    }
}

// ---------------------------------------------------------------------------
// Weight transpose (+ K zero-pad):  W[K,N] -> Wt[N,Kpad]
// ---------------------------------------------------------------------------
__global__ void transpose_pad_kernel(const float* __restrict__ W, float* __restrict__ Wt,
                                     int K, int N, int Kpad)
{
    int i = blockIdx.x * blockDim.x + threadIdx.x;
    if (i >= N * Kpad) return;
    int n = i / Kpad, kp = i - n * Kpad;
    Wt[i] = (kp < K) ? W[(size_t)kp * N + n] : 0.0f;
}

// edge_attr [E,16] -> [E,32] zero-padded
__global__ void pad_edge_kernel(const float* __restrict__ ea, float* __restrict__ epad)
{
    int i = blockIdx.x * blockDim.x + threadIdx.x;
    if (i >= EDGES * EPAD) return;
    int e = i >> 5, k = i & 31;
    epad[i] = (k < EDGE_DIM) ? ea[e * EDGE_DIM + k] : 0.0f;
}

// ---------------------------------------------------------------------------
// Fills
// ---------------------------------------------------------------------------
__global__ void fill_f32_kernel(float* p, float v, int n) {
    int i = blockIdx.x * blockDim.x + threadIdx.x;
    if (i < n) p[i] = v;
}
__global__ void fill_u8_kernel(unsigned char* p, unsigned char v, int n) {
    int i = blockIdx.x * blockDim.x + threadIdx.x;
    if (i < n) p[i] = v;
}

// ---------------------------------------------------------------------------
// Edge attention: one wave per (edge, head)
// ---------------------------------------------------------------------------
__global__ void __launch_bounds__(128)
edge_alpha_kernel(const float* __restrict__ q, const float* __restrict__ k,
                  const float* __restrict__ eemb, const int* __restrict__ src,
                  const int* __restrict__ dst, const unsigned char* __restrict__ em,
                  float* __restrict__ alpha, float* __restrict__ mx, int E)
{
    int wid = (blockIdx.x * blockDim.x + threadIdx.x) >> 5;
    int e = wid >> 2, h = wid & 3;
    if (e >= E) return;
    int lane = threadIdx.x & 31;
    int s = src[e], d = dst[e];
    const float* qp = q    + (size_t)d * HC + h * EMB;
    const float* kp = k    + (size_t)s * HC + h * EMB;
    const float* ep = eemb + (size_t)e * HC + h * EMB;
    float acc = 0.0f;
#pragma unroll
    for (int j = 0; j < 4; ++j) {
        int c = lane + 32 * j;
        acc += qp[c] * (kp[c] + ep[c]);
    }
    acc = wave_sum32(acc) * 0.08838834764831845f;   // 1/sqrt(128)
    if (lane == 0) {
        alpha[e * 4 + h] = acc;
        if (em[e]) atomicMaxF(&mx[d * 4 + h], acc);
    }
}

__global__ void edge_softmax_kernel(const float* __restrict__ alpha,
                                    const float* __restrict__ mx,
                                    const int* __restrict__ dst,
                                    const unsigned char* __restrict__ em,
                                    float* __restrict__ p, float* __restrict__ denom, int E)
{
    int i = blockIdx.x * blockDim.x + threadIdx.x;
    if (i >= E * 4) return;
    int e = i >> 2, h = i & 3;
    float pv = 0.0f;
    if (em[e]) {
        float m = mx[dst[e] * 4 + h];
        if (!(m > -1e37f)) m = 0.0f;                // isfinite fix
        pv = expf(alpha[i] - m);
    }
    p[i] = pv;
    if (pv != 0.0f) atomicAdd(&denom[dst[e] * 4 + h], pv);
}

__global__ void __launch_bounds__(128)
edge_aggregate_kernel(const float* __restrict__ p, const float* __restrict__ denom,
                      const float* __restrict__ v, const float* __restrict__ eemb,
                      const int* __restrict__ src, const int* __restrict__ dst,
                      const unsigned char* __restrict__ em, float* __restrict__ outagg, int E)
{
    int wid = (blockIdx.x * blockDim.x + threadIdx.x) >> 5;
    int e = wid >> 2, h = wid & 3;
    if (e >= E) return;
    if (!em[e]) return;
    int lane = threadIdx.x & 31;
    int s = src[e], d = dst[e];
    float wgt = p[e * 4 + h] / (denom[d * 4 + h] + 1e-16f);
    const float* vp = v    + (size_t)s * HC + h * EMB;
    const float* ep = eemb + (size_t)e * HC + h * EMB;
    float*       op = outagg + (size_t)d * HC + h * EMB;
#pragma unroll
    for (int j = 0; j < 4; ++j) {
        int c = lane + 32 * j;
        atomicAdd(&op[c], wgt * (vp[c] + ep[c]));
    }
}

// beta = sigmoid(cat(out,skip,out-skip)·Wb);  x = beta*skip + (1-beta)*out
__global__ void __launch_bounds__(128)
beta_blend_kernel(const float* __restrict__ outagg, const float* __restrict__ skip,
                  const float* __restrict__ Wb, float* __restrict__ xout, int Nn)
{
    int wid = (blockIdx.x * blockDim.x + threadIdx.x) >> 5;
    if (wid >= Nn) return;
    int lane = threadIdx.x & 31;
    const float* op = outagg + (size_t)wid * HC;
    const float* sp = skip   + (size_t)wid * HC;
    float acc = 0.0f;
#pragma unroll
    for (int j = 0; j < 16; ++j) {
        int c = lane + 32 * j;
        float o = op[c], s = sp[c];
        acc += o * Wb[c] + s * Wb[HC + c] + (o - s) * Wb[2 * HC + c];
    }
    acc = wave_sum32(acc);                         // all lanes hold full sum
    float beta = 1.0f / (1.0f + expf(-acc));
    float* xp = xout + (size_t)wid * HC;
#pragma unroll
    for (int j = 0; j < 16; ++j) {
        int c = lane + 32 * j;
        xp[c] = beta * sp[c] + (1.0f - beta) * op[c];
    }
}

// ---------------------------------------------------------------------------
// BatchNorm (training-mode biased stats). One block per channel.
// ---------------------------------------------------------------------------
__global__ void __launch_bounds__(256)
bn_stats_kernel(const float* __restrict__ x, float* mean, float* var, int M)
{
    int c = blockIdx.x;
    __shared__ float s1[256], s2[256];
    float a = 0.0f, b = 0.0f;
    for (int r = threadIdx.x; r < M; r += 256) {
        float v = x[(size_t)r * EMB + c];
        a += v; b += v * v;
    }
    s1[threadIdx.x] = a; s2[threadIdx.x] = b;
    __syncthreads();
    for (int off = 128; off > 0; off >>= 1) {
        if (threadIdx.x < off) {
            s1[threadIdx.x] += s1[threadIdx.x + off];
            s2[threadIdx.x] += s2[threadIdx.x + off];
        }
        __syncthreads();
    }
    if (threadIdx.x == 0) {
        float m = s1[0] / (float)M;
        mean[c] = m;
        var[c]  = s2[0] / (float)M - m * m;
    }
}

__global__ void bn_apply_kernel(const float* __restrict__ x, const float* __restrict__ mean,
                                const float* __restrict__ var, const float* __restrict__ g,
                                const float* __restrict__ b, float* __restrict__ y, int total)
{
    int i = blockIdx.x * blockDim.x + threadIdx.x;
    if (i >= total) return;
    int c = i & (EMB - 1);
    y[i] = g[c] * (x[i] - mean[c]) * rsqrtf(var[c] + 1e-5f) + b[c];
}

// ---------------------------------------------------------------------------
// TopKPooling: one block (128 threads) per graph.
// ---------------------------------------------------------------------------
__global__ void __launch_bounds__(128)
topk_pool_kernel(const float* __restrict__ x, const float* __restrict__ w,
                 float* __restrict__ xnew, unsigned char* __restrict__ kept,
                 int* __restrict__ new_id, int n, int kk)
{
    int g = blockIdx.x, tid = threadIdx.x;
    __shared__ float sc[NODES0];
    __shared__ unsigned char kp[NODES0];
    __shared__ float wn[EMB];
    wn[tid] = w[tid] * w[tid];
    __syncthreads();
    for (int off = 64; off > 0; off >>= 1) {
        if (tid < off) wn[tid] += wn[tid + off];
        __syncthreads();
    }
    float nrm = sqrtf(wn[0]);

    float si = -3.4e38f;
    if (tid < n) {
        const float* xp = x + (size_t)(g * n + tid) * EMB;
        float d = 0.0f;
        for (int c = 0; c < EMB; ++c) d += xp[c] * w[c];
        si = d / nrm;
        sc[tid] = si;
    }
    __syncthreads();

    int keep = 0;
    if (tid < n) {
        int r = 0;
        for (int j = 0; j < n; ++j) {
            float sj = sc[j];
            r += (sj > si) || (sj == si && j < tid);
        }
        keep = (r < kk);
    }
    kp[tid] = (unsigned char)keep;
    __syncthreads();

    if (tid < n) {
        int gi = g * n + tid;
        if (keep) {
            int pos = 0;
            for (int j = 0; j < tid; ++j) pos += kp[j];
            int ni = g * kk + pos;                 // global index in new numbering
            float t = tanhf(si);
            const float* xp = x + (size_t)gi * EMB;
            float*       yp = xnew + (size_t)ni * EMB;
            for (int c = 0; c < EMB; ++c) yp[c] = xp[c] * t;
            kept[gi]   = 1;
            new_id[gi] = ni;
        } else {
            kept[gi]   = 0;
            new_id[gi] = 0;
        }
    }
}

__global__ void edge_remap_kernel(const int* __restrict__ so, const int* __restrict__ dto,
                                  const unsigned char* __restrict__ emo,
                                  const unsigned char* __restrict__ kept,
                                  const int* __restrict__ nid,
                                  int* __restrict__ sn, int* __restrict__ dn,
                                  unsigned char* __restrict__ emn, int E)
{
    int e = blockIdx.x * blockDim.x + threadIdx.x;
    if (e >= E) return;
    int s = so[e], d = dto[e];
    int m = emo[e] && kept[s] && kept[d];
    sn[e]  = m ? nid[s] : 0;
    dn[e]  = m ? nid[d] : 0;
    emn[e] = (unsigned char)m;
}

// gmp || gap per graph; optional accumulate across layers
__global__ void __launch_bounds__(128)
graph_pool_kernel(const float* __restrict__ x, float* __restrict__ reps, int n, int accum)
{
    int g = blockIdx.x, c = threadIdx.x;
    float mx = -3.4e38f, s = 0.0f;
    for (int i = 0; i < n; ++i) {
        float v = x[(size_t)(g * n + i) * EMB + c];
        mx = fmaxf(mx, v);
        s += v;
    }
    float* rp = reps + (size_t)g * (2 * EMB);
    if (accum) { rp[c] += mx; rp[EMB + c] += s / (float)n; }
    else       { rp[c]  = mx; rp[EMB + c]  = s / (float)n; }
}

// final [rows,K] @ [K,1] + b
__global__ void __launch_bounds__(128)
rowdot_kernel(const float* __restrict__ h, const float* __restrict__ w,
              const float* __restrict__ b, float* __restrict__ out, int rows, int K)
{
    int wid = (blockIdx.x * blockDim.x + threadIdx.x) >> 5;
    if (wid >= rows) return;
    int lane = threadIdx.x & 31;
    float acc = 0.0f;
    for (int c = lane; c < K; c += 32) acc += h[(size_t)wid * K + c] * w[c];
    acc = wave_sum32(acc);
    if (lane == 0) out[wid] = acc + b[0];
}

// ---------------------------------------------------------------------------
// Host-side orchestration
// ---------------------------------------------------------------------------
static inline void run_transpose(hipStream_t st, const float* W, float* Wt,
                                 int K, int N, int Kpad) {
    int total = N * Kpad;
    transpose_pad_kernel<<<(total + 255) / 256, 256, 0, st>>>(W, Wt, K, N, Kpad);
}
static inline void run_gemm(hipStream_t st, const float* A, const float* Wt,
                            const float* bias, float* C, int M, int N, int K, int relu) {
    dim3 grid(N / 64, M / 64), block(128);
    wmma_gemm_nt_kernel<<<grid, block, 0, st>>>(A, Wt, bias, C, M, N, K, relu);
}
static inline void fill_f32(hipStream_t st, float* p, float v, int n) {
    fill_f32_kernel<<<(n + 255) / 256, 256, 0, st>>>(p, v, n);
}

struct TWs {
    float *q, *k, *v, *skip, *eemb, *alpha, *p, *mx, *denom, *outagg;
    float *wtq, *wtk, *wtv, *wts, *wte;   // transposed weights
};

static void run_tconv(hipStream_t st, const float* x, int F, int Nn,
                      const int* src, const int* dst, const unsigned char* em,
                      const float* Wq, const float* bq, const float* Wk, const float* bk,
                      const float* Wv, const float* bv, const float* We,
                      const float* Wsk, const float* bs, const float* Wb,
                      const float* epad, const TWs& w, float* xout)
{
    // transpose weights -> [N,K] for the NT GEMM
    run_transpose(st, Wq,  w.wtq, F, HC, F);
    run_transpose(st, Wk,  w.wtk, F, HC, F);
    run_transpose(st, Wv,  w.wtv, F, HC, F);
    run_transpose(st, Wsk, w.wts, F, HC, F);
    run_transpose(st, We,  w.wte, EDGE_DIM, HC, EPAD);

    run_gemm(st, x, w.wtq, bq, w.q,    Nn, HC, F, 0);
    run_gemm(st, x, w.wtk, bk, w.k,    Nn, HC, F, 0);
    run_gemm(st, x, w.wtv, bv, w.v,    Nn, HC, F, 0);
    run_gemm(st, x, w.wts, bs, w.skip, Nn, HC, F, 0);
    run_gemm(st, epad, w.wte, nullptr, w.eemb, EDGES, HC, EPAD, 0);

    fill_f32(st, w.mx,     -INFINITY, Nn * HEADS);
    fill_f32(st, w.denom,  0.0f,      Nn * HEADS);
    fill_f32(st, w.outagg, 0.0f,      Nn * HC);

    edge_alpha_kernel<<<EDGES * HEADS / 4, 128, 0, st>>>(w.q, w.k, w.eemb, src, dst, em,
                                                         w.alpha, w.mx, EDGES);
    edge_softmax_kernel<<<EDGES * HEADS / 256, 256, 0, st>>>(w.alpha, w.mx, dst, em,
                                                             w.p, w.denom, EDGES);
    edge_aggregate_kernel<<<EDGES * HEADS / 4, 128, 0, st>>>(w.p, w.denom, w.v, w.eemb,
                                                             src, dst, em, w.outagg, EDGES);
    beta_blend_kernel<<<Nn / 4, 128, 0, st>>>(w.outagg, w.skip, Wb, xout, Nn);
}

extern "C" void kernel_launch(void* const* d_in, const int* in_sizes, int n_in,
                              void* d_out, int out_size, void* d_ws, size_t ws_size,
                              hipStream_t stream)
{
    (void)in_sizes; (void)n_in; (void)out_size; (void)ws_size;

    // ---- inputs (setup_inputs dict order) ----
    const float* x_in  = (const float*)d_in[0];
    const int*   eidx  = (const int*)  d_in[1];
    const float* eattr = (const float*)d_in[2];
    const float* c1_Wq = (const float*)d_in[3];  const float* c1_bq = (const float*)d_in[4];
    const float* c1_Wk = (const float*)d_in[5];  const float* c1_bk = (const float*)d_in[6];
    const float* c1_Wv = (const float*)d_in[7];  const float* c1_bv = (const float*)d_in[8];
    const float* c1_We = (const float*)d_in[9];
    const float* c1_Ws = (const float*)d_in[10]; const float* c1_bs = (const float*)d_in[11];
    const float* c1_Wb = (const float*)d_in[12];
    const float* t1_W  = (const float*)d_in[13]; const float* t1_b  = (const float*)d_in[14];
    const float* bn1_g = (const float*)d_in[15]; const float* bn1_b = (const float*)d_in[16];
    const float* cl_Wq = (const float*)d_in[17]; const float* cl_bq = (const float*)d_in[18];
    const float* cl_Wk = (const float*)d_in[19]; const float* cl_bk = (const float*)d_in[20];
    const float* cl_Wv = (const float*)d_in[21]; const float* cl_bv = (const float*)d_in[22];
    const float* cl_We = (const float*)d_in[23];
    const float* cl_Ws = (const float*)d_in[24]; const float* cl_bs = (const float*)d_in[25];
    const float* cl_Wb = (const float*)d_in[26];
    const float* tl_W  = (const float*)d_in[27]; const float* tl_b  = (const float*)d_in[28];
    const float* bnl_g = (const float*)d_in[29]; const float* bnl_b = (const float*)d_in[30];
    const float* pool_w = (const float*)d_in[31];
    const float* l1_W  = (const float*)d_in[32]; const float* l1_b  = (const float*)d_in[33];
    const float* l2_W  = (const float*)d_in[34]; const float* l2_b  = (const float*)d_in[35];
    const float* l3_W  = (const float*)d_in[36]; const float* l3_b  = (const float*)d_in[37];

    const int* src0 = eidx;
    const int* dst0 = eidx + EDGES;

    // ---- workspace carve ----
    char* bp = (char*)d_ws;
    auto alloc = [&](size_t bytes) -> char* {
        char* r = bp; bp += (bytes + 255) & ~(size_t)255; return r;
    };
    TWs tw;
    tw.q      = (float*)alloc((size_t)NMAX * HC * 4);
    tw.k      = (float*)alloc((size_t)NMAX * HC * 4);
    tw.v      = (float*)alloc((size_t)NMAX * HC * 4);
    tw.skip   = (float*)alloc((size_t)NMAX * HC * 4);
    tw.outagg = (float*)alloc((size_t)NMAX * HC * 4);
    tw.eemb   = (float*)alloc((size_t)EDGES * HC * 4);
    tw.alpha  = (float*)alloc((size_t)EDGES * HEADS * 4);
    tw.p      = (float*)alloc((size_t)EDGES * HEADS * 4);
    tw.mx     = (float*)alloc((size_t)NMAX * HEADS * 4);
    tw.denom  = (float*)alloc((size_t)NMAX * HEADS * 4);
    tw.wtq    = (float*)alloc((size_t)HC * EMB * 4);   // [512, F<=128]
    tw.wtk    = (float*)alloc((size_t)HC * EMB * 4);
    tw.wtv    = (float*)alloc((size_t)HC * EMB * 4);
    tw.wts    = (float*)alloc((size_t)HC * EMB * 4);
    tw.wte    = (float*)alloc((size_t)HC * EPAD * 4);  // [512, 32]

    float* epad   = (float*)alloc((size_t)EDGES * EPAD * 4);   // padded edge_attr
    float* wtt    = (float*)alloc((size_t)EMB * HC * 4);       // [128, 512] t1/tl transposed
    float* wl1t   = (float*)alloc((size_t)DENSE * 2 * EMB * 4);
    float* wl2t   = (float*)alloc((size_t)(DENSE / 2) * DENSE * 4);
    float* xblend = (float*)alloc((size_t)NMAX * HC * 4);      // tconv output [Nn, 512]
    float* xa     = (float*)alloc((size_t)NMAX * EMB * 4);     // pre-BN
    float* xb     = (float*)alloc((size_t)NMAX * EMB * 4);     // post-BN
    float* xpool1 = (float*)alloc((size_t)(NMAX / 2) * EMB * 4);
    float* xpool2 = (float*)alloc((size_t)(NMAX / 4) * EMB * 4);
    float* mean   = (float*)alloc(EMB * 4);
    float* var    = (float*)alloc(EMB * 4);
    unsigned char* em0  = (unsigned char*)alloc(EDGES);
    unsigned char* em1  = (unsigned char*)alloc(EDGES);
    unsigned char* kept = (unsigned char*)alloc(NMAX);
    int* nid  = (int*)alloc((size_t)NMAX * 4);
    int* src1 = (int*)alloc((size_t)EDGES * 4);
    int* dst1 = (int*)alloc((size_t)EDGES * 4);
    float* reps = (float*)alloc((size_t)NGRAPH * 2 * EMB * 4);
    float* h1   = (float*)alloc((size_t)NGRAPH * DENSE * 4);
    float* h2   = (float*)alloc((size_t)NGRAPH * (DENSE / 2) * 4);

    // ---- graph setup ----
    fill_u8_kernel<<<EDGES / 256, 256, 0, stream>>>(em0, 1, EDGES);
    pad_edge_kernel<<<EDGES * EPAD / 256, 256, 0, stream>>>(eattr, epad);

    // ---- conv1 (F_IN=64) + transf1 + bn1 ----
    run_tconv(stream, x_in, F_IN, NMAX, src0, dst0, em0,
              c1_Wq, c1_bq, c1_Wk, c1_bk, c1_Wv, c1_bv, c1_We, c1_Ws, c1_bs, c1_Wb,
              epad, tw, xblend);
    run_transpose(stream, t1_W, wtt, HC, EMB, HC);
    run_gemm(stream, xblend, wtt, t1_b, xa, NMAX, EMB, HC, 1);        // relu fused
    bn_stats_kernel<<<EMB, 256, 0, stream>>>(xa, mean, var, NMAX);
    bn_apply_kernel<<<(NMAX * EMB + 255) / 256, 256, 0, stream>>>(xa, mean, var,
                                                                  bn1_g, bn1_b, xb, NMAX * EMB);

    // ---- layer 0: tconv + transf + bn + topk-pool + graph-pool ----
    run_tconv(stream, xb, EMB, NMAX, src0, dst0, em0,
              cl_Wq, cl_bq, cl_Wk, cl_bk, cl_Wv, cl_bv, cl_We, cl_Ws, cl_bs, cl_Wb,
              epad, tw, xblend);
    run_transpose(stream, tl_W, wtt, HC, EMB, HC);
    run_gemm(stream, xblend, wtt, tl_b, xa, NMAX, EMB, HC, 1);
    bn_stats_kernel<<<EMB, 256, 0, stream>>>(xa, mean, var, NMAX);
    bn_apply_kernel<<<(NMAX * EMB + 255) / 256, 256, 0, stream>>>(xa, mean, var,
                                                                  bnl_g, bnl_b, xb, NMAX * EMB);
    topk_pool_kernel<<<NGRAPH, 128, 0, stream>>>(xb, pool_w, xpool1, kept, nid,
                                                 NODES0, NODES0 / 2);
    edge_remap_kernel<<<EDGES / 256, 256, 0, stream>>>(src0, dst0, em0, kept, nid,
                                                       src1, dst1, em1, EDGES);
    graph_pool_kernel<<<NGRAPH, 128, 0, stream>>>(xpool1, reps, NODES0 / 2, 0);

    // ---- layer 1 (Nn = 8192, n = 64) ----
    const int N1 = NMAX / 2, n1 = NODES0 / 2;
    run_tconv(stream, xpool1, EMB, N1, src1, dst1, em1,
              cl_Wq + (size_t)EMB * HC, cl_bq + HC,
              cl_Wk + (size_t)EMB * HC, cl_bk + HC,
              cl_Wv + (size_t)EMB * HC, cl_bv + HC,
              cl_We + (size_t)EDGE_DIM * HC,
              cl_Ws + (size_t)EMB * HC, cl_bs + HC,
              cl_Wb + 3 * HC, epad, tw, xblend);
    run_transpose(stream, tl_W + (size_t)HC * EMB, wtt, HC, EMB, HC);
    run_gemm(stream, xblend, wtt, tl_b + EMB, xa, N1, EMB, HC, 1);
    bn_stats_kernel<<<EMB, 256, 0, stream>>>(xa, mean, var, N1);
    bn_apply_kernel<<<(N1 * EMB + 255) / 256, 256, 0, stream>>>(xa, mean, var,
                                                                bnl_g + EMB, bnl_b + EMB,
                                                                xb, N1 * EMB);
    topk_pool_kernel<<<NGRAPH, 128, 0, stream>>>(xb, pool_w + EMB, xpool2, kept, nid,
                                                 n1, n1 / 2);
    graph_pool_kernel<<<NGRAPH, 128, 0, stream>>>(xpool2, reps, n1 / 2, 1); // accumulate

    // ---- MLP head ----
    run_transpose(stream, l1_W, wl1t, 2 * EMB, DENSE, 2 * EMB);
    run_gemm(stream, reps, wl1t, l1_b, h1, NGRAPH, DENSE, 2 * EMB, 1);
    run_transpose(stream, l2_W, wl2t, DENSE, DENSE / 2, DENSE);
    run_gemm(stream, h1, wl2t, l2_b, h2, NGRAPH, DENSE / 2, DENSE, 1);
    rowdot_kernel<<<NGRAPH / 4, 128, 0, stream>>>(h2, l3_W, l3_b, (float*)d_out,
                                                  NGRAPH, DENSE / 2);
}
